// EdgeConvBlock_82085414961804
// MI455X (gfx1250) — matrix operand order repaired and checked by reading, not compiled, not run on hardware
//
#include <hip/hip_runtime.h>
#include <stdint.h>

// ---------------------------------------------------------------------------
// EdgeConv block for MI455X (gfx1250, wave32, WMMA).
// Pipeline:
//   knn -> featsT -> x0(edge feats, bf16) -> [GEMM(bf16 wmma) -> BN stats]*3
//   -> shortcut GEMM -> BN stats -> fused mean_k + add + relu + transpose.
// BN+ReLU of layer i is folded into the A-operand load of layer i+1.
// ---------------------------------------------------------------------------

#define B_   32
#define N_   1024
#define CIN_ 64
#define K_   16
#define C_   128           // all three layer widths
#define EPS_ 1e-5f

typedef __attribute__((ext_vector_type(16))) __bf16 v16bf;
typedef __attribute__((ext_vector_type(8)))  float  v8f;

union U16x16 { v16bf v; uint32_t u[8]; };

static __device__ __forceinline__ float bfbits2f(uint16_t u) {
    uint32_t x = ((uint32_t)u) << 16;
    float f; __builtin_memcpy(&f, &x, 4); return f;
}
static __device__ __forceinline__ uint16_t f2bfbits(float f) {
    uint32_t x; __builtin_memcpy(&x, &f, 4);
    uint32_t r = (x + 0x7fffu + ((x >> 16) & 1u)) >> 16;   // RNE
    return (uint16_t)r;
}

// ---------------------------------------------------------------------------
// 1) kNN: one thread per query point; 1024 points staged in LDS.
//    Keeps 16 nearest (self excluded) via unrolled insertion network.
// ---------------------------------------------------------------------------
__global__ __launch_bounds__(128) void knn_kernel(const float* __restrict__ points,
                                                  int* __restrict__ idx) {
    __shared__ float px[N_], py[N_], pz[N_];
    const int b  = blockIdx.x >> 3;          // 8 blocks of 128 queries per batch
    const int n0 = (blockIdx.x & 7) * 128;
    const float* P = points + (long)b * 3 * N_;
    for (int i = threadIdx.x; i < N_; i += 128) {
        px[i] = P[i]; py[i] = P[N_ + i]; pz[i] = P[2 * N_ + i];
    }
    __syncthreads();
    const int n = n0 + threadIdx.x;
    const float qx = px[n], qy = py[n], qz = pz[n];
    float bd[K_]; int bi[K_];
#pragma unroll
    for (int i = 0; i < K_; ++i) { bd[i] = 3.4e38f; bi[i] = 0; }
    for (int j = 0; j < N_; ++j) {
        float dx = px[j] - qx, dy = py[j] - qy, dz = pz[j] - qz;
        float d2 = dx * dx + dy * dy + dz * dz;
        if (j != n && d2 < bd[K_ - 1]) {
            bd[K_ - 1] = d2; bi[K_ - 1] = j;
#pragma unroll
            for (int s = K_ - 1; s > 0; --s) {
                if (bd[s] < bd[s - 1]) {
                    float td = bd[s]; bd[s] = bd[s - 1]; bd[s - 1] = td;
                    int   ti = bi[s]; bi[s] = bi[s - 1]; bi[s - 1] = ti;
                }
            }
        }
    }
    int* o = idx + ((long)b * N_ + n) * K_;
#pragma unroll
    for (int k = 0; k < K_; ++k) o[k] = bi[k];
}

// ---------------------------------------------------------------------------
// 2) features (B,CIN,N) -> featsT fp32 + bf16, row-major (B*N, CIN)
// ---------------------------------------------------------------------------
__global__ void transpose_feats(const float* __restrict__ F, float* __restrict__ T32,
                                uint16_t* __restrict__ T16, int total) {
    int t = blockIdx.x * blockDim.x + threadIdx.x;
    if (t >= total) return;
    int n = t & (N_ - 1); int c = (t >> 10) & (CIN_ - 1); int b = t >> 16;
    float v = F[t];
    long o = ((long)b * N_ + n) * CIN_ + c;
    T32[o] = v; T16[o] = f2bfbits(v);
}

__global__ void f32_to_bf16(const float* __restrict__ s, uint16_t* __restrict__ d, int n) {
    int t = blockIdx.x * blockDim.x + threadIdx.x;
    if (t < n) d[t] = f2bfbits(s[t]);
}

// ---------------------------------------------------------------------------
// 3) edge features x0[(b,n,k), 0:128] = [center(64) | nbr - center(64)] (bf16)
// ---------------------------------------------------------------------------
__global__ __launch_bounds__(128) void build_x0(const float* __restrict__ featsT,
                                                const int* __restrict__ idx,
                                                uint16_t* __restrict__ X0) {
    const int q = blockIdx.x;                 // b*N + n
    const int t = threadIdx.x;                // 128
    const int c = t & (CIN_ - 1);
    const int half = t >> 6;
    const int b = q >> 10;
    const float center = featsT[(long)q * CIN_ + c];
    for (int k = 0; k < K_; ++k) {
        long row = (long)q * K_ + k;
        float v;
        if (half == 0) v = center;
        else {
            int j = idx[q * K_ + k];
            v = featsT[((long)b * N_ + j) * CIN_ + c] - center;
        }
        X0[row * C_ + half * CIN_ + c] = f2bfbits(v);
    }
}

// ---------------------------------------------------------------------------
// 4) GEMM: Y(M x 128 bf16) = act(A) (M x Kdim bf16) @ W^T (Kdim x 128 bf16)
//    act = use_aff ? relu(a*scale+shift) : identity (folded prev-layer BN+ReLU)
//    256 threads = 8 waves; wave owns 16 rows x 128 cols (8 wmma C-tiles).
//    Operand layouts used here (contiguous K, same for A and B; wave32, 16-bit, 16x16x32):
//      A (16x32): lane&15 = row; lanes<16 hold K 0..15 contiguous,
//                 lanes>=16 hold K 16..31 contiguous (pairs per VGPR).
//      B (32x16): lane&15 = column; lanes<16 hold K 0..15 contiguous,
//                 lanes>=16 hold K 16..31 contiguous (row striped per VGPR).
// ---------------------------------------------------------------------------
__global__ __launch_bounds__(256) void gemm_wmma(const uint16_t* __restrict__ A,
                                                 const uint16_t* __restrict__ W,
                                                 uint16_t* __restrict__ Y,
                                                 const float* __restrict__ aff,
                                                 int Kdim, int use_aff) {
    const int lane = threadIdx.x & 31;
    const int wave = threadIdx.x >> 5;
    const int ln   = lane & 15;
    const int hi   = lane >> 4;
    const long rowBase = (long)blockIdx.x * 128 + wave * 16;
    const uint16_t* Arow = A + (rowBase + ln) * (long)Kdim;

    v8f c[8] = {};

    for (int kb = 0; kb < Kdim; kb += 32) {
        // ---- A operand (16 contiguous K per lane-half, matches B) ----
        U16x16 a;
#pragma unroll
        for (int j = 0; j < 8; ++j) {
            const int kp = kb + 2 * j + hi * 16;
            uint32_t pk; __builtin_memcpy(&pk, Arow + kp, 4);
            if (use_aff) {
                float lo = bfbits2f((uint16_t)pk);
                float hf = bfbits2f((uint16_t)(pk >> 16));
                lo = fmaxf(fmaf(lo, aff[kp],     aff[128 + kp]),     0.f);
                hf = fmaxf(fmaf(hf, aff[kp + 1], aff[128 + kp + 1]), 0.f);
                pk = (uint32_t)f2bfbits(lo) | ((uint32_t)f2bfbits(hf) << 16);
            }
            a.u[j] = pk;
        }
        // ---- 8 N-tiles; B operand = 16 contiguous K per lane ----
#pragma unroll
        for (int nt = 0; nt < 8; ++nt) {
            const uint16_t* Wp = W + (nt * 16 + ln) * (long)Kdim + kb + hi * 16;
            U16x16 bm;
#pragma unroll
            for (int j = 0; j < 8; ++j) {
                __builtin_memcpy(&bm.u[j], Wp + 2 * j, 4);
            }
            c[nt] = __builtin_amdgcn_wmma_f32_16x16x32_bf16(
                false, a.v, false, bm.v, (short)0, c[nt], false, false);
        }
    }
    // C/D layout: VGPR i -> row i + 8*hi, lane (ln) -> column
#pragma unroll
    for (int nt = 0; nt < 8; ++nt) {
#pragma unroll
        for (int i = 0; i < 8; ++i) {
            long r = rowBase + i + hi * 8;
            Y[r * C_ + nt * 16 + ln] = f2bfbits(c[nt][i]);
        }
    }
}

// ---------------------------------------------------------------------------
// 5) BN statistics, deterministic two-stage reduction over rows.
// ---------------------------------------------------------------------------
__global__ __launch_bounds__(256) void stats_partial(const uint16_t* __restrict__ Ybf,
                                                     float* __restrict__ partials,
                                                     int M, int rowsPerBlock) {
    __shared__ float ssum[256], ssq[256];
    const int tid = threadIdx.x;
    const int c = tid & 127, half = tid >> 7;
    long r0 = (long)blockIdx.x * rowsPerBlock + half;
    long r1 = (long)(blockIdx.x + 1) * rowsPerBlock;
    if (r1 > M) r1 = M;
    float s = 0.f, q = 0.f;
    for (long r = r0; r < r1; r += 2) {
        float v = bfbits2f(Ybf[r * C_ + c]);
        s += v; q += v * v;
    }
    ssum[tid] = s; ssq[tid] = q;
    __syncthreads();
    if (half == 0) {
        partials[blockIdx.x * 256 + c]       = ssum[c] + ssum[c + 128];
        partials[blockIdx.x * 256 + 128 + c] = ssq[c]  + ssq[c + 128];
    }
}

__global__ __launch_bounds__(128) void stats_final(const float* __restrict__ partials,
                                                   int nb, float invM,
                                                   const float* __restrict__ gamma,
                                                   const float* __restrict__ beta,
                                                   float* __restrict__ affine) {
    const int c = threadIdx.x;
    double s = 0.0, q = 0.0;
    for (int i = 0; i < nb; ++i) {
        s += (double)partials[i * 256 + c];
        q += (double)partials[i * 256 + 128 + c];
    }
    float mean = (float)(s * (double)invM);
    float var  = (float)(q * (double)invM) - mean * mean;
    float rstd = rsqrtf(var + EPS_);
    float sc = gamma[c] * rstd;
    affine[c] = sc;
    affine[128 + c] = beta[c] - mean * sc;
}

// ---------------------------------------------------------------------------
// 6) out[b,o,n] = relu( bn_sc(sc[b,n,o]) + mean_k relu(bn2(y2[b,n,k,o])) )
// ---------------------------------------------------------------------------
__global__ __launch_bounds__(128) void final_combine(const uint16_t* __restrict__ Y2,
                                                     const uint16_t* __restrict__ SC,
                                                     const float* __restrict__ aff2,
                                                     const float* __restrict__ affsc,
                                                     float* __restrict__ out) {
    const int q = blockIdx.x;                 // b*N + n
    const int o = threadIdx.x;                // 128
    const int b = q >> 10, n = q & (N_ - 1);
    const float s2 = aff2[o], t2 = aff2[128 + o];
    float acc = 0.f;
#pragma unroll
    for (int k = 0; k < K_; ++k) {
        float v = bfbits2f(Y2[((long)q * K_ + k) * C_ + o]);
        acc += fmaxf(fmaf(v, s2, t2), 0.f);
    }
    float fts = acc * (1.f / (float)K_);
    float sv = bfbits2f(SC[(long)q * C_ + o]);
    sv = fmaf(sv, affsc[o], affsc[128 + o]);
    out[(long)b * C_ * N_ + (long)o * N_ + n] = fmaxf(sv + fts, 0.f);
}

// ---------------------------------------------------------------------------
// Host-side launcher
// ---------------------------------------------------------------------------
extern "C" void kernel_launch(void* const* d_in, const int* in_sizes, int n_in,
                              void* d_out, int out_size, void* d_ws, size_t ws_size,
                              hipStream_t stream) {
    (void)in_sizes; (void)n_in; (void)out_size; (void)ws_size;
    const float* points   = (const float*)d_in[0];
    const float* features = (const float*)d_in[1];
    const float* W0  = (const float*)d_in[2];
    const float* g0  = (const float*)d_in[3];
    const float* b0  = (const float*)d_in[4];
    const float* W1  = (const float*)d_in[5];
    const float* g1  = (const float*)d_in[6];
    const float* b1  = (const float*)d_in[7];
    const float* W2  = (const float*)d_in[8];
    const float* g2  = (const float*)d_in[9];
    const float* b2  = (const float*)d_in[10];
    const float* Wsc = (const float*)d_in[11];
    const float* gsc = (const float*)d_in[12];
    const float* bsc = (const float*)d_in[13];
    float* out = (float*)d_out;

    const long BN  = (long)B_ * N_;          // 32768
    const long M   = BN * K_;                // 524288 rows for the 3 MLP GEMMs

    // ---- carve workspace (~292 MB; aligned 256B) ----
    char* p = (char*)d_ws;
    auto carve = [&](size_t bytes) {
        char* r = p;
        p += (bytes + 255) & ~(size_t)255;
        return r;
    };
    int*      idx    = (int*)     carve(BN * K_ * sizeof(int));            // 2 MB
    float*    fT32   = (float*)   carve(BN * CIN_ * sizeof(float));        // 8 MB
    uint16_t* fT16   = (uint16_t*)carve(BN * CIN_ * sizeof(uint16_t));     // 4 MB
    uint16_t* buf0   = (uint16_t*)carve((size_t)M * C_ * 2);               // 128 MB
    uint16_t* buf1   = (uint16_t*)carve((size_t)M * C_ * 2);               // 128 MB
    uint16_t* scbuf  = (uint16_t*)carve((size_t)BN * C_ * 2);              // 8 MB
    uint16_t* W0b    = (uint16_t*)carve(C_ * C_ * 2);
    uint16_t* W1b    = (uint16_t*)carve(C_ * C_ * 2);
    uint16_t* W2b    = (uint16_t*)carve(C_ * C_ * 2);
    uint16_t* Wscb   = (uint16_t*)carve(C_ * CIN_ * 2);
    float*    part   = (float*)   carve(256 * 256 * sizeof(float));
    float*    aff0   = (float*)   carve(256 * sizeof(float));
    float*    aff1   = (float*)   carve(256 * sizeof(float));
    float*    aff2   = (float*)   carve(256 * sizeof(float));
    float*    affsc  = (float*)   carve(256 * sizeof(float));

    // 1) kNN
    knn_kernel<<<B_ * 8, 128, 0, stream>>>(points, idx);

    // 2) transpose features + bf16 weight conversion
    {
        int total = (int)(BN * CIN_);
        transpose_feats<<<(total + 255) / 256, 256, 0, stream>>>(features, fT32, fT16, total);
        f32_to_bf16<<<(C_ * C_ + 255) / 256, 256, 0, stream>>>(W0, W0b, C_ * C_);
        f32_to_bf16<<<(C_ * C_ + 255) / 256, 256, 0, stream>>>(W1, W1b, C_ * C_);
        f32_to_bf16<<<(C_ * C_ + 255) / 256, 256, 0, stream>>>(W2, W2b, C_ * C_);
        f32_to_bf16<<<(C_ * CIN_ + 255) / 256, 256, 0, stream>>>(Wsc, Wscb, C_ * CIN_);
    }

    // 3) edge features
    build_x0<<<(int)BN, 128, 0, stream>>>(fT32, idx, buf0);

    const int RPB = 2048;
    const int nbM  = (int)(M / RPB);     // 256
    const int nbBN = (int)(BN / RPB);    // 16

    // 4) layer 0: buf0 -> buf1 ; stats -> aff0
    gemm_wmma<<<(int)(M / 128), 256, 0, stream>>>(buf0, W0b, buf1, aff0, C_, 0);
    stats_partial<<<nbM, 256, 0, stream>>>(buf1, part, (int)M, RPB);
    stats_final<<<1, 128, 0, stream>>>(part, nbM, 1.0f / (float)M, g0, b0, aff0);

    // 5) layer 1: act(buf1; aff0) -> buf0 ; stats -> aff1
    gemm_wmma<<<(int)(M / 128), 256, 0, stream>>>(buf1, W1b, buf0, aff0, C_, 1);
    stats_partial<<<nbM, 256, 0, stream>>>(buf0, part, (int)M, RPB);
    stats_final<<<1, 128, 0, stream>>>(part, nbM, 1.0f / (float)M, g1, b1, aff1);

    // 6) layer 2: act(buf0; aff1) -> buf1 ; stats -> aff2
    gemm_wmma<<<(int)(M / 128), 256, 0, stream>>>(buf0, W2b, buf1, aff1, C_, 1);
    stats_partial<<<nbM, 256, 0, stream>>>(buf1, part, (int)M, RPB);
    stats_final<<<1, 128, 0, stream>>>(part, nbM, 1.0f / (float)M, g2, b2, aff2);

    // 7) shortcut: featsT -> scbuf ; stats -> affsc
    gemm_wmma<<<(int)(BN / 128), 256, 0, stream>>>(fT16, Wscb, scbuf, affsc, CIN_, 0);
    stats_partial<<<nbBN, 256, 0, stream>>>(scbuf, part, (int)BN, RPB);
    stats_final<<<1, 128, 0, stream>>>(part, nbBN, 1.0f / (float)BN, gsc, bsc, affsc);

    // 8) fused BN2+ReLU+mean_k + BNsc + add + ReLU + transpose
    final_combine<<<(int)BN, 128, 0, stream>>>(buf1, scbuf, aff2, affsc, out);
}
